// RecurrentEncoder_37675453120651
// MI455X (gfx1250) — matrix operand order repaired and checked by reading, not compiled, not run on hardware
//
#include <hip/hip_runtime.h>

#define NND 100000
#define FN  16
#define FD  48
#define TT  24
#define HH  256
#define KIN 64      // FN + FD
#define NT  64      // nodes per workgroup (4 row-blocks of 16)
#define NWG 1563    // ceil(100000 / 64)
#define LDA 264     // LDS row stride (ushorts) for 256-wide bf16 tiles (+8 pad)
#define LDX 72      // LDS row stride for 64-wide X tile (+8 pad)
#define CP  260     // LDS row stride (floats) for c-state (+4 pad)

typedef __attribute__((ext_vector_type(16))) __bf16        bf16x16;
typedef __attribute__((ext_vector_type(8)))  float         f32x8;
typedef __attribute__((ext_vector_type(4)))  unsigned int  u32x4;

union Frag { bf16x16 v; u32x4 q[2]; };

__device__ __forceinline__ unsigned short f2bf(float f) {
  __bf16 b = (__bf16)f;                       // RNE, lowers to v_cvt_pk_bf16_f32
  return __builtin_bit_cast(unsigned short, b);
}
__device__ __forceinline__ float tanh_(float x) {
#if __has_builtin(__builtin_amdgcn_tanhf)
  return __builtin_amdgcn_tanhf(x);           // native V_TANH_F32 (CDNA5 trans op)
#elif __has_builtin(__builtin_amdgcn_tanh_f32)
  return __builtin_amdgcn_tanh_f32(x);
#else
  float e = __expf(-2.0f * x);
  return (1.0f - e) / (1.0f + e);
#endif
}
__device__ __forceinline__ float sigmoid_(float x) {
  // sigmoid(x) = 0.5*tanh(x/2) + 0.5  -> one trans op + one FMA
  return __builtin_fmaf(0.5f, tanh_(0.5f * x), 0.5f);
}
__device__ __forceinline__ f32x8 wmma_bf16(bf16x16 a, bf16x16 b, f32x8 c) {
  return __builtin_amdgcn_wmma_f32_16x16x32_bf16(false, a, false, b, (short)0, c, false, false);
}
// A fragment (16x32 bf16): lane holds M=lane%16, K=(lane/16)*16+{0..15}
__device__ __forceinline__ bf16x16 load_a(const unsigned short* p) {
  Frag f; f.q[0] = *(const u32x4*)p; f.q[1] = *(const u32x4*)(p + 8); return f.v;
}
// B fragment (32x16 bf16): lane holds N=lane%16, K=(lane/16)*16+{0..15}; for gates = x@W^T with
// row-major W this is 16 CONTIGUOUS elements of W[n0+N] starting at k0+(lane/16)*16.
__device__ __forceinline__ bf16x16 load_b(const unsigned short* p) {
  Frag f; f.q[0] = *(const u32x4*)p; f.q[1] = *(const u32x4*)(p + 8); return f.v;
}

__global__ void repack_bf16(const float* __restrict__ src, unsigned short* __restrict__ dst, int n) {
  int i = blockIdx.x * blockDim.x + threadIdx.x;
  if (i < n) dst[i] = f2bf(src[i]);
}

// One K-pass of a cell GEMM: 4 B fragments (one per gate) reused across 4 row-blocks.
// Weight address = SGPR base + (wOff + k0) + constant-immediate gate offset.
#define GEMM_PART(SA, W)                                                     \
  _Pragma("unroll 1")                                                        \
  for (int k0 = 0; k0 < HH; k0 += 32) {                                      \
    bf16x16 b0 = load_b(W + wOff + k0);                                      \
    bf16x16 b1 = load_b(W + wOff + k0 + 1 * (256 * HH));                     \
    bf16x16 b2 = load_b(W + wOff + k0 + 2 * (256 * HH));                     \
    bf16x16 b3 = load_b(W + wOff + k0 + 3 * (256 * HH));                     \
    _Pragma("unroll")                                                        \
    for (int rb = 0; rb < 4; ++rb) {                                         \
      bf16x16 a = load_a(&SA[rb * 16 * LDA + aRow + k0]);                    \
      acc[rb * 4 + 0] = wmma_bf16(a, b0, acc[rb * 4 + 0]);                   \
      acc[rb * 4 + 1] = wmma_bf16(a, b1, acc[rb * 4 + 1]);                   \
      acc[rb * 4 + 2] = wmma_bf16(a, b2, acc[rb * 4 + 2]);                   \
      acc[rb * 4 + 3] = wmma_bf16(a, b3, acc[rb * 4 + 3]);                   \
    }                                                                        \
  }

// Elementwise LSTM update: c lives in LDS, h written straight to LDS (caller barriers first).
#define CELL_ELEM(CLDS, HLDS, PH, PC, LAST)                                  \
  _Pragma("unroll")                                                          \
  for (int rb = 0; rb < 4; ++rb) {                                           \
    _Pragma("unroll")                                                        \
    for (int v = 0; v < 8; ++v) {                                            \
      int r = rb * 16 + v + 8 * khalf;                                       \
      float ig = sigmoid_(acc[rb * 4 + 0][v]);                               \
      float fg = sigmoid_(acc[rb * 4 + 1][v]);                               \
      float gg = tanh_   (acc[rb * 4 + 2][v]);                               \
      float og = sigmoid_(acc[rb * 4 + 3][v]);                               \
      float c  = fg * CLDS[r * CP + jCol] + ig * gg;                         \
      CLDS[r * CP + jCol] = c;                                               \
      float h  = og * tanh_(c);                                              \
      HLDS[r * LDA + jCol] = f2bf(h);                                        \
      if (LAST) {                                                            \
        int n = n0 + r;                                                      \
        if (n < NND) {                                                       \
          size_t o = (size_t)n * HH + jCol;                                  \
          out[(size_t)(PH) * NND * HH + o] = h;                              \
          out[(size_t)(PC) * NND * HH + o] = c;                              \
        }                                                                    \
      }                                                                      \
    }                                                                        \
  }

__global__ __launch_bounds__(512)
void lstm_persistent(const float* __restrict__ node,
                     const float* __restrict__ dyn,
                     const float* __restrict__ b_ih0, const float* __restrict__ b_hh0,
                     const float* __restrict__ b_ih1, const float* __restrict__ b_hh1,
                     const unsigned short* __restrict__ Win16,
                     const unsigned short* __restrict__ Wih0,
                     const unsigned short* __restrict__ Whh0,
                     const unsigned short* __restrict__ Wih1,
                     const unsigned short* __restrict__ Whh1,
                     float* __restrict__ out) {
  __shared__ __align__(16) unsigned short sX  [NT * LDX];   // [64][64] concat(node, dyn_t)
  __shared__ __align__(16) unsigned short sInp[NT * LDA];   // input projection (bf16)
  __shared__ __align__(16) unsigned short sH0 [NT * LDA];   // layer-0 h (bf16)
  __shared__ __align__(16) unsigned short sH1 [NT * LDA];   // layer-1 h (bf16)
  __shared__ __align__(16) float          sC0 [NT * CP];    // layer-0 c (f32)
  __shared__ __align__(16) float          sC1 [NT * CP];    // layer-1 c (f32)

  const int tid   = threadIdx.x;
  const int lane  = tid & 31;
  const int wv    = tid >> 5;        // wave 0..15 -> hidden units [16*wv, 16*wv+16)
  const int col   = lane & 15;
  const int khalf = lane >> 4;
  const int kh16  = khalf * 16;
  const int n0    = blockIdx.x * NT;
  const int jCol  = wv * 16 + col;                    // this lane's hidden column

  // ---- one-time tile setup ----
  for (int i = tid; i < NT * FN; i += 512) {          // static node columns of X
    int m = i >> 4, c = i & 15;
    int g = (n0 + m < NND) ? (n0 + m) : (NND - 1);
    sX[m * LDX + c] = f2bf(node[(size_t)g * FN + c]);
  }
  for (int i = tid; i < NT * LDA; i += 512) { sH0[i] = 0; sH1[i] = 0; }
  for (int i = tid; i < NT * CP;  i += 512) { sC0[i] = 0.0f; sC1[i] = 0.0f; }

  float bias0[4], bias1[4];
#pragma unroll
  for (int u = 0; u < 4; ++u) {                       // gates ordered i,f,g,o
    int nn   = u * HH + jCol;
    bias0[u] = b_ih0[nn] + b_hh0[nn];
    bias1[u] = b_ih1[nn] + b_hh1[nn];
  }

  const int aRow  = col * LDA + kh16;                 // A-fragment base, 256-wide LDS tiles
  const int aRowX = col * LDX + kh16;                 // A-fragment base, X tile
  const int wOff  = jCol * HH + kh16;                 // per-lane B offset, 1024x256 weights
  const int wOffX = jCol * KIN + kh16;                // per-lane B offset, 256x64 W_in

  __syncthreads();

  for (int step = 0; step < TT; ++step) {
    // gather this step's dyn features straight from global into X (cols 16..63);
    // each (n,f) cacheline is reused across all 24 steps of this WG's lifetime.
    for (int p = tid; p < NT * FD; p += 512) {
      int m = p / FD, f = p % FD;
      int g = (n0 + m < NND) ? (n0 + m) : (NND - 1);
      sX[m * LDX + FN + f] = f2bf(dyn[(size_t)g * (FD * TT) + f * TT + step]);
    }
    __syncthreads();                                   // X ready

    { // ---- input projection: inp = X @ W_in^T (no bias); 1 col tile x 4 row blocks ----
      f32x8 pz = {0, 0, 0, 0, 0, 0, 0, 0};
      f32x8 p[4] = {pz, pz, pz, pz};
#pragma unroll 1
      for (int k0 = 0; k0 < KIN; k0 += 32) {
        bf16x16 b = load_b(Win16 + wOffX + k0);
#pragma unroll
        for (int rb = 0; rb < 4; ++rb)
          p[rb] = wmma_bf16(load_a(&sX[rb * 16 * LDX + aRowX + k0]), b, p[rb]);
      }
#pragma unroll
      for (int rb = 0; rb < 4; ++rb)
#pragma unroll
        for (int v = 0; v < 8; ++v)
          sInp[(rb * 16 + v + 8 * khalf) * LDA + jCol] = f2bf(p[rb][v]);
    }
    __syncthreads();                                   // inp visible to all waves

    f32x8 acc[16];
    const int last = (step == TT - 1);

    // ---- layer 0 ----
#pragma unroll
    for (int u = 0; u < 4; ++u) {
      float b = bias0[u];
      f32x8 bb = {b, b, b, b, b, b, b, b};
#pragma unroll
      for (int rb = 0; rb < 4; ++rb) acc[rb * 4 + u] = bb;
    }
    GEMM_PART(sInp, Wih0);
    GEMM_PART(sH0, Whh0);
    __syncthreads();                                   // all GEMM reads of old h0 done
    CELL_ELEM(sC0, sH0, 0, 1, last);
    __syncthreads();                                   // new h0 visible

    // ---- layer 1 (input = new h0) ----
#pragma unroll
    for (int u = 0; u < 4; ++u) {
      float b = bias1[u];
      f32x8 bb = {b, b, b, b, b, b, b, b};
#pragma unroll
      for (int rb = 0; rb < 4; ++rb) acc[rb * 4 + u] = bb;
    }
    GEMM_PART(sH0, Wih1);
    GEMM_PART(sH1, Whh1);
    __syncthreads();                                   // all GEMM reads of old h1 done
    CELL_ELEM(sC1, sH1, 2, 3, last);
    __syncthreads();                                   // new h1 visible
  }
}

extern "C" void kernel_launch(void* const* d_in, const int* in_sizes, int n_in,
                              void* d_out, int out_size, void* d_ws, size_t ws_size,
                              hipStream_t stream) {
  (void)in_sizes; (void)n_in; (void)out_size; (void)ws_size;
  const float* node = (const float*)d_in[0];
  const float* dyn  = (const float*)d_in[1];
  const float* Win  = (const float*)d_in[2];
  const float* wih0 = (const float*)d_in[3];
  const float* whh0 = (const float*)d_in[4];
  const float* bih0 = (const float*)d_in[5];
  const float* bhh0 = (const float*)d_in[6];
  const float* wih1 = (const float*)d_in[7];
  const float* whh1 = (const float*)d_in[8];
  const float* bih1 = (const float*)d_in[9];
  const float* bhh1 = (const float*)d_in[10];
  float* out = (float*)d_out;

  // d_ws layout (bf16/ushort): W_in [256*64] | w_ih0 | w_hh0 | w_ih1 | w_hh1 [1024*256 each]
  unsigned short* ws    = (unsigned short*)d_ws;
  unsigned short* Win16 = ws;
  unsigned short* W0i   = ws + 16384;
  unsigned short* W0h   = W0i + 262144;
  unsigned short* W1i   = W0h + 262144;
  unsigned short* W1h   = W1i + 262144;

  repack_bf16<<<64,   256, 0, stream>>>(Win,  Win16, 16384);
  repack_bf16<<<1024, 256, 0, stream>>>(wih0, W0i,   262144);
  repack_bf16<<<1024, 256, 0, stream>>>(whh0, W0h,   262144);
  repack_bf16<<<1024, 256, 0, stream>>>(wih1, W1i,   262144);
  repack_bf16<<<1024, 256, 0, stream>>>(whh1, W1h,   262144);

  lstm_persistent<<<NWG, 512, 0, stream>>>(
      node, dyn, bih0, bhh0, bih1, bhh1, Win16, W0i, W0h, W1i, W1h, out);
}
